// CompositeOutputHead_24790551232587
// MI455X (gfx1250) — matrix operand order repaired and checked by reading, not compiled, not run on hardware
//
#include <hip/hip_runtime.h>

// ---------------------------------------------------------------------------
// Chunked cross-entropy head for MI455X (gfx1250, wave32, WMMA).
//   x: (4,2048,1024) f32   w: (50257,1024) f32   targets:(4,2048) i32
//   mask:(4,2048) bool(u8)  chunk_tokens: scalar (mathematically irrelevant)
// Output: [loss, n_correct, n_valid] as f32.
//
// 843 GFLOP GEMM -> compute bound. bf16 hi/lo split (3 WMMA passes) gives
// fp32-class accuracy at bf16 matrix-pipe speed. Pre-pass swizzles x and w
// into matching contiguous-k WMMA A/B fragment layouts so the hot loop is only b128 loads
// + v_wmma_f32_16x16x32_bf16 with a 2x2 register macro-tile (12 WMMA per
// 16 loads). Softmax stats are kept per-lane (no shuffles in the hot loop);
// cross-lane/cross-wave merges happen once at the end. Fully deterministic
// (no atomics).
// ---------------------------------------------------------------------------

typedef __attribute__((ext_vector_type(16))) __bf16 v16bf;
typedef __attribute__((ext_vector_type(8)))  float  v8f;

#define D_DIM   1024
#define V_DIM   50257
#define N_TOK   8192
#define V_TILES ((V_DIM + 15) / 16)   // 3142 vocab tiles of 16 (even)
#define V_PAIRS (V_TILES / 2)         // 1571 vocab tile-pairs
#define R_TILES (N_TOK / 16)          // 512 token row-tiles of 16
#define M_BLK   64                    // tokens per block
#define NBLOCKS (N_TOK / M_BLK)       // 128

// Workspace layout (bytes). Requires ws_size >= ~240 MB.
static constexpr unsigned long long XSZ    = (unsigned long long)R_TILES * 16384ull * 2ull; // 16 MB
static constexpr unsigned long long WSZ    = (unsigned long long)V_TILES * 16384ull * 2ull; // ~98 MB
static constexpr unsigned long long XH_OFF = 4096ull;
static constexpr unsigned long long XL_OFF = XH_OFF + XSZ;
static constexpr unsigned long long WH_OFF = XL_OFF + XSZ;
static constexpr unsigned long long WL_OFF = WH_OFF + WSZ;

// ---------------------------------------------------------------------------
// Pre-pass: x (f32) -> bf16 hi/lo in a contiguous-k A-fragment layout (same k order as w).
// A 16x32 bf16: lane<16: row=lane, elems[0..15]=K k0+0..15
//               lane>=16: row=lane-16, elems = K k0+16..31
// chunk = ((rt*32 + kk)*32 + lane); 16 bf16 (32B) contiguous per chunk.
// ---------------------------------------------------------------------------
__global__ void cvt_x_kernel(const float* __restrict__ x,
                             __bf16* __restrict__ xh, __bf16* __restrict__ xl) {
  unsigned chunk = blockIdx.x * blockDim.x + threadIdx.x;
  if (chunk >= (unsigned)R_TILES * 1024u) return;
  unsigned rt  = chunk >> 10, rem = chunk & 1023u;
  unsigned kk  = rem >> 5,   l   = rem & 31u;
  unsigned row = l & 15u, off8 = (l >> 4) * 16u, k0 = kk * 32u;
  const float* src = x + (size_t)(rt * 16u + row) * D_DIM + k0;
  v16bf hi, lo;
#pragma unroll
  for (int j = 0; j < 8; ++j) {
    float a = src[off8 + j];
    __bf16 ha = (__bf16)a;
    hi[j] = ha; lo[j] = (__bf16)(a - (float)ha);
    float b = src[8u + off8 + j];
    __bf16 hb = (__bf16)b;
    hi[8 + j] = hb; lo[8 + j] = (__bf16)(b - (float)hb);
  }
  *(v16bf*)(xh + (size_t)chunk * 16) = hi;
  *(v16bf*)(xl + (size_t)chunk * 16) = lo;
}

// ---------------------------------------------------------------------------
// Pre-pass: w (f32) -> bf16 hi/lo in a contiguous-k B-fragment layout (same k order as x).
// B 32x16 bf16: lane<16: col=lane, elems[0..15]=K k0+0..15
//               lane>=16: col=lane-16, elems = K k0+16..31
// Padding rows (v >= V) written as zeros (masked to -3e38 in main kernel).
// ---------------------------------------------------------------------------
__global__ void cvt_w_kernel(const float* __restrict__ w,
                             __bf16* __restrict__ wh, __bf16* __restrict__ wl) {
  unsigned chunk = blockIdx.x * blockDim.x + threadIdx.x;
  if (chunk >= (unsigned)V_TILES * 1024u) return;
  unsigned t  = chunk >> 10, rem = chunk & 1023u;
  unsigned kk = rem >> 5,   l   = rem & 31u;
  unsigned v  = t * 16u + (l & 15u);
  unsigned k0 = kk * 32u + (l >> 4) * 16u;
  v16bf hi, lo;
  if (v < V_DIM) {
    const float* src = w + (size_t)v * D_DIM + k0;
#pragma unroll
    for (int j = 0; j < 16; ++j) {
      float a = src[j];
      __bf16 h = (__bf16)a;
      hi[j] = h; lo[j] = (__bf16)(a - (float)h);
    }
  } else {
#pragma unroll
    for (int j = 0; j < 16; ++j) { hi[j] = (__bf16)0.0f; lo[j] = (__bf16)0.0f; }
  }
  *(v16bf*)(wh + (size_t)chunk * 16) = hi;
  *(v16bf*)(wl + (size_t)chunk * 16) = lo;
}

// ---------------------------------------------------------------------------
// Main kernel: 128 blocks x 64 tokens. 8 waves = 2 row-pairs x 4 vocab
// streams; each wave computes a 32x32 logit macro-tile per iteration
// (MT=2 x NT=2 register blocking): per K-step 16 b128 loads, 12 WMMAs.
// Lane-local online softmax/argmax/target-capture; merges deferred.
// ---------------------------------------------------------------------------
__global__ __launch_bounds__(256) void ce_main_kernel(
    const int* __restrict__ targets, const unsigned char* __restrict__ mask,
    const __bf16* __restrict__ xh, const __bf16* __restrict__ xl,
    const __bf16* __restrict__ wh, const __bf16* __restrict__ wl,
    float* __restrict__ p_loss, int* __restrict__ p_corr, int* __restrict__ p_valid) {
  const float NEG = -3.0e38f;
  const int tid  = threadIdx.x;
  const int wave = tid >> 5, lane = tid & 31;
  const int mg = wave >> 2;                  // row-pair: row tiles {2mg, 2mg+1}
  const int nw = wave & 3;                   // vocab stream (0..3)
  const int rt0 = blockIdx.x * 4 + mg * 2;
  const int hl = lane >> 4, col = lane & 15; // C/D layout: col=lane&15, row=r+8*hl

  // Per-lane state for 16 rows: idx = i*8+r -> token (rt0+i)*16 + r + 8*hl.
  int   tgt[16];
  float rmax[16], rsum[16], amv[16], tgv[16];
  int   ami[16];
#pragma unroll
  for (int q = 0; q < 16; ++q) {
    int tok = (rt0 + (q >> 3)) * 16 + (q & 7) + 8 * hl;
    int tv = targets[tok];
    tgt[q]  = tv < 0 ? 0 : tv;               // clamp(min=0) like the reference
    rmax[q] = NEG; rsum[q] = 0.0f; amv[q] = NEG; ami[q] = 0x7fffffff; tgv[q] = 0.0f;
  }

  const size_t xoff0 = ((size_t)rt0 * 1024 + lane) * 16;
  const size_t xoff1 = xoff0 + 16384;        // next row tile

  for (int tp = nw; tp < V_PAIRS; tp += 4) {
    const int t0 = 2 * tp, t1 = 2 * tp + 1;
    const size_t woff0 = ((size_t)t0 * 1024 + lane) * 16;
    const size_t woff1 = woff0 + 16384;
    v8f a00 = {0.f,0.f,0.f,0.f,0.f,0.f,0.f,0.f};
    v8f a01 = a00, a10 = a00, a11 = a00;
#pragma unroll 4
    for (int kk = 0; kk < 32; ++kk) {
      const size_t ko = (size_t)kk * 512;
      v16bf ah0 = *(const v16bf*)(xh + xoff0 + ko);
      v16bf al0 = *(const v16bf*)(xl + xoff0 + ko);
      v16bf ah1 = *(const v16bf*)(xh + xoff1 + ko);
      v16bf al1 = *(const v16bf*)(xl + xoff1 + ko);
      v16bf bh0 = *(const v16bf*)(wh + woff0 + ko);
      v16bf bl0 = *(const v16bf*)(wl + woff0 + ko);
      v16bf bh1 = *(const v16bf*)(wh + woff1 + ko);
      v16bf bl1 = *(const v16bf*)(wl + woff1 + ko);
      a00 = __builtin_amdgcn_wmma_f32_16x16x32_bf16(false, ah0, false, bh0, (short)0, a00, false, false);
      a01 = __builtin_amdgcn_wmma_f32_16x16x32_bf16(false, ah0, false, bh1, (short)0, a01, false, false);
      a10 = __builtin_amdgcn_wmma_f32_16x16x32_bf16(false, ah1, false, bh0, (short)0, a10, false, false);
      a11 = __builtin_amdgcn_wmma_f32_16x16x32_bf16(false, ah1, false, bh1, (short)0, a11, false, false);
      a00 = __builtin_amdgcn_wmma_f32_16x16x32_bf16(false, ah0, false, bl0, (short)0, a00, false, false);
      a01 = __builtin_amdgcn_wmma_f32_16x16x32_bf16(false, ah0, false, bl1, (short)0, a01, false, false);
      a10 = __builtin_amdgcn_wmma_f32_16x16x32_bf16(false, ah1, false, bl0, (short)0, a10, false, false);
      a11 = __builtin_amdgcn_wmma_f32_16x16x32_bf16(false, ah1, false, bl1, (short)0, a11, false, false);
      a00 = __builtin_amdgcn_wmma_f32_16x16x32_bf16(false, al0, false, bh0, (short)0, a00, false, false);
      a01 = __builtin_amdgcn_wmma_f32_16x16x32_bf16(false, al0, false, bh1, (short)0, a01, false, false);
      a10 = __builtin_amdgcn_wmma_f32_16x16x32_bf16(false, al1, false, bh0, (short)0, a10, false, false);
      a11 = __builtin_amdgcn_wmma_f32_16x16x32_bf16(false, al1, false, bh1, (short)0, a11, false, false);
    }
    // Lane-local online softmax / argmax / target capture (no shuffles here).
    const int cg0 = t0 * 16 + col, cg1 = t1 * 16 + col;
#pragma unroll
    for (int q4 = 0; q4 < 4; ++q4) {
      const int i = q4 >> 1, j = q4 & 1;
      const int cg = j ? cg1 : cg0;
      const v8f& ac = i ? (j ? a11 : a10) : (j ? a01 : a00);
#pragma unroll
      for (int r = 0; r < 8; ++r) {
        const int q = i * 8 + r;
        float v = (cg < V_DIM) ? ac[r] : NEG;  // pad columns -> -inf-ish
        if (cg == tgt[q]) tgv[q] = v;          // target < V always
        float Mn = fmaxf(rmax[q], v);
        rsum[q] = rsum[q] * __expf(rmax[q] - Mn) + __expf(v - Mn);
        rmax[q] = Mn;
        if (v > amv[q] || (v == amv[q] && cg < ami[q])) { amv[q] = v; ami[q] = cg; }
      }
    }
  }

  // One-time cross-lane merge over the 16-lane column group.
  // xor masks <= 8 never cross the 16-lane halves, so both halves (rows r and
  // r+8) reduce independently with full-width shuffles.
#pragma unroll
  for (int q = 0; q < 16; ++q) {
#pragma unroll
    for (int off = 1; off <= 8; off <<= 1) {
      float om = __shfl_xor(rmax[q], off);
      float os = __shfl_xor(rsum[q], off);
      float M  = fmaxf(rmax[q], om);
      rsum[q] = rsum[q] * __expf(rmax[q] - M) + os * __expf(om - M);
      rmax[q] = M;
      float av = __shfl_xor(amv[q], off);
      int   ai = __shfl_xor(ami[q], off);
      if (av > amv[q] || (av == amv[q] && ai < ami[q])) { amv[q] = av; ami[q] = ai; }
      tgv[q] += __shfl_xor(tgv[q], off);
    }
  }

  __shared__ float smax[4][64], ssum[4][64], samv[4][64], stgt[4][64];
  __shared__ int   sami[4][64];
  if (col == 0) {                              // lanes 0 (rows r) and 16 (rows r+8)
#pragma unroll
    for (int q = 0; q < 16; ++q) {
      int row = mg * 32 + (q >> 3) * 16 + 8 * hl + (q & 7);
      smax[nw][row] = rmax[q]; ssum[nw][row] = rsum[q];
      samv[nw][row] = amv[q];  sami[nw][row] = ami[q];
      stgt[nw][row] = tgv[q];
    }
  }
  __syncthreads();

  __shared__ float sloss[64];
  __shared__ int   scorr[64], svalid[64];
  if (tid < 64) {
    const int row = tid, token = blockIdx.x * 64 + row;
    float M = smax[0][row], S = ssum[0][row], tg = stgt[0][row];
    float av = samv[0][row]; int ai = sami[0][row];
#pragma unroll
    for (int s = 1; s < 4; ++s) {              // fixed-order merge (deterministic)
      float m2 = smax[s][row];
      float Mn = fmaxf(M, m2);
      S  = S * __expf(M - Mn) + ssum[s][row] * __expf(m2 - Mn);
      M  = Mn;
      tg += stgt[s][row];
      float av2 = samv[s][row]; int ai2 = sami[s][row];
      if (av2 > av || (av2 == av && ai2 < ai)) { av = av2; ai = ai2; }
    }
    float lse = M + __logf(S);
    int tv = targets[token]; int tc = tv < 0 ? 0 : tv;
    bool mk = mask[token] != 0;
    sloss[row]  = mk ? (lse - tg) : 0.0f;      // raw CE = logsumexp - target logit
    scorr[row]  = (mk && ai == tc) ? 1 : 0;
    svalid[row] = mk ? 1 : 0;
  }
  __syncthreads();
  if (tid == 0) {                              // fixed-order in-block reduce
    float ls = 0.0f; int cc = 0, vv = 0;
    for (int i = 0; i < 64; ++i) { ls += sloss[i]; cc += scorr[i]; vv += svalid[i]; }
    p_loss[blockIdx.x] = ls; p_corr[blockIdx.x] = cc; p_valid[blockIdx.x] = vv;
  }
}

// Deterministic final reduction (no atomics anywhere -> bitwise-stable replays).
__global__ void finalize_kernel(const float* __restrict__ pl, const int* __restrict__ pc,
                                const int* __restrict__ pv, float* __restrict__ out) {
  if (blockIdx.x == 0 && threadIdx.x == 0) {
    float ls = 0.0f; int cc = 0, vv = 0;
    for (int i = 0; i < NBLOCKS; ++i) { ls += pl[i]; cc += pc[i]; vv += pv[i]; }
    int dv = vv < 1 ? 1 : vv;
    out[0] = ls / (float)dv;
    out[1] = (float)cc;
    out[2] = (float)vv;
  }
}

extern "C" void kernel_launch(void* const* d_in, const int* in_sizes, int n_in,
                              void* d_out, int out_size, void* d_ws, size_t ws_size,
                              hipStream_t stream) {
  (void)in_sizes; (void)n_in; (void)out_size; (void)ws_size;
  const float*         x       = (const float*)d_in[0];
  const float*         w       = (const float*)d_in[1];
  const int*           targets = (const int*)d_in[2];
  const unsigned char* mask    = (const unsigned char*)d_in[3];  // jnp bool = u8
  // d_in[4] = chunk_tokens: chunking does not change the result; ignored.

  char* ws = (char*)d_ws;
  float*  p_loss  = (float*)(ws);
  int*    p_corr  = (int*)(ws + 512);
  int*    p_valid = (int*)(ws + 1024);
  __bf16* xh = (__bf16*)(ws + XH_OFF);
  __bf16* xl = (__bf16*)(ws + XL_OFF);
  __bf16* wh = (__bf16*)(ws + WH_OFF);
  __bf16* wl = (__bf16*)(ws + WL_OFF);
  float* outp = (float*)d_out;

  cvt_x_kernel<<<(R_TILES * 1024 + 255) / 256, 256, 0, stream>>>(x, xh, xl);
  cvt_w_kernel<<<(V_TILES * 1024 + 255) / 256, 256, 0, stream>>>(w, wh, wl);
  ce_main_kernel<<<NBLOCKS, 256, 0, stream>>>(targets, mask, xh, xl, wh, wl,
                                              p_loss, p_corr, p_valid);
  finalize_kernel<<<1, 1, 0, stream>>>(p_loss, p_corr, p_valid, outp);
}
